// GATModel_3461743641201
// MI455X (gfx1250) — matrix operand order repaired and checked by reading, not compiled, not run on hardware
//
#include <hip/hip_runtime.h>
#include <hip/hip_bf16.h>
#include <math.h>

// ---------------------------------------------------------------------------
// 2-layer GAT for MI455X (gfx1250, wave32).
//   - GEMMs via v_wmma_f32_16x16x32_bf16, with B pre-packed in the WMMA
//     register layout (coalesced b128 loads) and A staged in LDS as
//     pre-swizzled bf16 (ds_load_b128, no per-element cvt in the hot loop).
//     WMMA kernel is branch-free: full 16-row tiles only; remainder rows
//     (N%16) go through a tiny scalar tail kernel.
//   - Exact top-k threshold via 3-level radix select on float bit patterns.
//   - Segment softmax via native CDNA5 float atomics
//     (global_atomic_max_num_f32 / global_atomic_add_f32).
//   - Edge kernels templated on (heads, channels) -> shifts instead of idiv.
// ---------------------------------------------------------------------------

typedef __attribute__((ext_vector_type(16))) __bf16 v16bf;
typedef __attribute__((ext_vector_type(8)))  float  v8f;

#define IN_F   128
#define HIDC   32
#define HEADS  4
#define OUT_F  64
#define NEG_SLOPE 0.2f

__device__ __forceinline__ float leaky(float v) { return v >= 0.f ? v : NEG_SLOPE * v; }

__device__ __forceinline__ void atomic_fmax(float* p, float v) {
  __hip_atomic_fetch_max(p, v, __ATOMIC_RELAXED, __HIP_MEMORY_SCOPE_AGENT);
}
__device__ __forceinline__ void atomic_fadd(float* p, float v) {
  __hip_atomic_fetch_add(p, v, __ATOMIC_RELAXED, __HIP_MEMORY_SCOPE_AGENT);
}

// ---------------------------------------------------------------------------
// utility fills
// ---------------------------------------------------------------------------
__global__ void fill_f32(float* p, float v, int n) {
  int i = blockIdx.x * blockDim.x + threadIdx.x;
  if (i < n) p[i] = v;
}
__global__ void fill_u32(unsigned* p, unsigned v, int n) {
  int i = blockIdx.x * blockDim.x + threadIdx.x;
  if (i < n) p[i] = v;
}

// ---------------------------------------------------------------------------
// exact k-th largest edge weight: 3-level radix select (11 / 11 / 10 bits).
// edge weights are uniform [0,1) -> positive floats -> bit order == float order
// ---------------------------------------------------------------------------
__global__ void thresh_hist(const float* __restrict__ w, int E,
                            const unsigned* __restrict__ state,
                            unsigned* __restrict__ hist, int pass) {
  int i = blockIdx.x * blockDim.x + threadIdx.x;
  if (i >= E) return;
  unsigned bits = __float_as_uint(w[i]);
  if (pass == 0) {
    atomicAdd(&hist[bits >> 21], 1u);
  } else if (pass == 1) {
    if ((bits >> 21) == state[0]) atomicAdd(&hist[(bits >> 10) & 0x7FFu], 1u);
  } else {
    if ((bits >> 10) == state[0]) atomicAdd(&hist[bits & 0x3FFu], 1u);
  }
}

__global__ void thresh_scan(unsigned* hist, unsigned* state, int nbins, int pass,
                            unsigned k_init, float* thr) {
  if (threadIdx.x != 0 || blockIdx.x != 0) return;
  unsigned k = (pass == 0) ? k_init : state[1];
  unsigned acc = 0, sel = 0;
  for (int b = nbins - 1; b >= 0; --b) {
    unsigned c = hist[b];
    if (acc + c >= k) { sel = (unsigned)b; k -= acc; break; }
    acc += c;
  }
  state[1] = k;
  if (pass == 0)      state[0] = sel;
  else if (pass == 1) state[0] = (state[0] << 11) | sel;
  else                *thr = __uint_as_float((state[0] << 10) | sel);
  for (int b = 0; b < nbins; ++b) hist[b] = 0;   // ready for next pass
}

// ---------------------------------------------------------------------------
// Pack W[128 x NOUT] (f32, row major) into the WMMA B-operand register layout:
//   Wpack[((ks4*NT + tile)*32 + lane)*16 + j] = bf16( W[k][col] )
//   with col = tile*16 + (lane&15),  k = ks4*32 + (lane>>4)*8 + (j&7) + (j&8 ? 16 : 0)
// Each lane's 16 B values for one WMMA are then one contiguous 32B chunk.
// ---------------------------------------------------------------------------
template <int NOUT>
__global__ void pack_B(const float* __restrict__ W, __bf16* __restrict__ Wp) {
  constexpr int NT = NOUT / 16;
  int q = blockIdx.x * blockDim.x + threadIdx.x;
  if (q >= 4 * NT * 32 * 16) return;
  int j    = q & 15;
  int lane = (q >> 4) & 31;
  int tile = (q >> 9) % NT;
  int ks4  = q / (512 * NT);
  int col  = tile * 16 + (lane & 15);
  int k    = ks4 * 32 + (lane >> 4) * 8 + (j & 7) + ((j & 8) ? 16 : 0);
  Wp[q] = (__bf16)W[(size_t)k * NOUT + col];
}

// ---------------------------------------------------------------------------
// GEMM (full tiles only): H[16 rows x NOUT] per block = X_tile * Wpack
// One block = 16 output rows, NOUT/16 waves; each wave owns a 16x16 col tile.
// A tile converted to bf16 and pre-swizzled into the WMMA A layout in LDS.
// Hot loop: 2x ds_load_b128 (A) + 2x global_load_b128 (B) + 1 WMMA per K-step.
// Branch-free epilogue: base pointer + immediate-offset stores.
// ---------------------------------------------------------------------------
template <int NOUT>
__global__ void gemm_wmma(const float* __restrict__ X, const __bf16* __restrict__ Wp,
                          float* __restrict__ H) {
  constexpr int NT = NOUT / 16;
  __shared__ unsigned Al[4 * 32 * 8];      // 4 ksteps * 32 lanes * 16 bf16 = 4KB
  const int nthr    = 2 * NOUT;            // = blockDim.x
  const int rowbase = blockIdx.x * 16;

  // stage + convert + swizzle A: 1024 bf16-pairs (k is always even -> b64 loads)
  for (int p = threadIdx.x; p < 1024; p += nthr) {
    int ks4  = p >> 8;
    int lane = (p >> 3) & 31;
    int jp   = p & 7;                       // pair index: covers j=2jp, 2jp+1
    int rr   = lane & 15;
    int hf   = lane >> 4;
    int k    = ks4 * 32 + hf * 8 + (jp & 3) * 2 + ((jp & 4) ? 16 : 0);
    int gr   = rowbase + rr;
    float2 v = *(const float2*)&X[(size_t)gr * IN_F + k];
    union { __bf16 h[2]; unsigned u; } cv;
    cv.h[0] = (__bf16)v.x;
    cv.h[1] = (__bf16)v.y;
    Al[p] = cv.u;
  }
  __syncthreads();

  const int wave = threadIdx.x >> 5;
  const int lane = threadIdx.x & 31;
  const int r    = lane & 15;
  const int half = lane >> 4;
  const int col  = wave * 16 + r;

  v8f c = {};
#pragma unroll
  for (int ks4 = 0; ks4 < 4; ++ks4) {
    v16bf a = *(const v16bf*)&Al[(ks4 * 32 + lane) * 8];
    v16bf b = *(const v16bf*)&Wp[((size_t)(ks4 * NT + wave) * 32 + lane) * 16];
    c = __builtin_amdgcn_wmma_f32_16x16x32_bf16(false, a, false, b, (short)0, c,
                                                false, false);
  }

  // C/D layout: VGPR j -> M = j + 8*half, N = lane&15
  float* hp = H + (size_t)(rowbase + half * 8) * NOUT + col;
#pragma unroll
  for (int j = 0; j < 8; ++j) hp[j * NOUT] = c[j];
}

// scalar tail for the last N%16 rows (never launched when N%16==0)
template <int NOUT>
__global__ void gemm_tail(const float* __restrict__ X, const float* __restrict__ W,
                          float* __restrict__ H, int row0, int nrows) {
  int t = blockIdx.x * blockDim.x + threadIdx.x;
  if (t >= nrows * NOUT) return;
  int row = row0 + t / NOUT;
  int col = t % NOUT;
  float acc = 0.f;
#pragma unroll 4
  for (int k = 0; k < IN_F; ++k)
    acc += X[(size_t)row * IN_F + k] * W[(size_t)k * NOUT + col];
  H[(size_t)row * NOUT + col] = acc;
}

// ---------------------------------------------------------------------------
// attention logits: alpha_src[n,h] = h[n,h,:]·a_src[h,:], same for dst
// ---------------------------------------------------------------------------
template <int Hh, int C>
__global__ void alpha_kernel(const float* __restrict__ Hm,
                             const float* __restrict__ a_src,
                             const float* __restrict__ a_dst,
                             float* __restrict__ as, float* __restrict__ ad,
                             int Nn) {
  int t = blockIdx.x * blockDim.x + threadIdx.x;
  if (t >= Nn * Hh) return;
  int n = t / Hh, hh = t % Hh;               // Hh is a power of two -> shifts
  const float* hp = Hm + (size_t)n * Hh * C + (size_t)hh * C;
  float s = 0.f, d = 0.f;
#pragma unroll 4
  for (int c = 0; c < C; ++c) {
    float v = hp[c];
    s += v * a_src[hh * C + c];
    d += v * a_dst[hh * C + c];
  }
  as[t] = s;
  ad[t] = d;
}

// ---------------------------------------------------------------------------
// edge passes. Virtual edges e in [E, E+Nn) are self-loops (always valid).
// phase 0: segment max (global_atomic_max_num_f32)
// phase 1: softmax denominator (global_atomic_add_f32)
// ---------------------------------------------------------------------------
template <int Hh>
__global__ void edge_score_kernel(const int* __restrict__ src, const int* __restrict__ dst,
                                  const float* __restrict__ w, const float* __restrict__ thr,
                                  const float* __restrict__ as, const float* __restrict__ ad,
                                  float* __restrict__ mx, float* __restrict__ den,
                                  int E, int Nn, int phase) {
  int t = blockIdx.x * blockDim.x + threadIdx.x;
  if (t >= (E + Nn) * Hh) return;
  int e = t / Hh, hh = t % Hh;               // power of two -> shifts
  int s, d;
  if (e < E) {
    if (!(w[e] >= *thr)) return;
    s = src[e]; d = dst[e];
  } else {
    s = d = e - E;
  }
  float sc = leaky(as[s * Hh + hh] + ad[d * Hh + hh]);
  if (phase == 0) atomic_fmax(&mx[d * Hh + hh], sc);
  else            atomic_fadd(&den[d * Hh + hh], __expf(sc - mx[d * Hh + hh]));
}

// aggregate: one wave per edge; lanes stride the H*C feature channels
template <int Hh, int C>
__global__ void edge_agg_kernel(const int* __restrict__ src, const int* __restrict__ dst,
                                const float* __restrict__ w, const float* __restrict__ thr,
                                const float* __restrict__ as, const float* __restrict__ ad,
                                const float* __restrict__ mx, const float* __restrict__ den,
                                const float* __restrict__ Hm, float* __restrict__ agg,
                                int E, int Nn) {
  constexpr int F = Hh * C;
  int gt   = blockIdx.x * blockDim.x + threadIdx.x;
  int e    = gt >> 5;
  int lane = gt & 31;
  if (e >= E + Nn) return;
  int s, d;
  if (e < E) {
    if (!(w[e] >= *thr)) return;
    s = src[e]; d = dst[e];
  } else {
    s = d = e - E;
  }
  // overlap the random-access gather with the softmax-weight math
  __builtin_prefetch(&Hm[(size_t)s * F + lane], 0, 0);
#pragma unroll
  for (int f = lane; f < F; f += 32) {
    int hh = f / C;                           // constexpr C -> shift
    float sc = leaky(as[s * Hh + hh] + ad[d * Hh + hh]);
    float al = __expf(sc - mx[d * Hh + hh]) / den[d * Hh + hh];
    atomic_fadd(&agg[(size_t)d * F + f], Hm[(size_t)s * F + f] * al);
  }
}

__global__ void relu_bias_kernel(float* x, const float* __restrict__ b, int n, int F) {
  int i = blockIdx.x * blockDim.x + threadIdx.x;
  if (i >= n) return;
  float v = x[i] + b[i & (F - 1)];
  x[i] = v > 0.f ? v : 0.f;
}
__global__ void add_bias_kernel(float* x, const float* __restrict__ b, int n, int F) {
  int i = blockIdx.x * blockDim.x + threadIdx.x;
  if (i >= n) return;
  x[i] += b[i & (F - 1)];
}

// ---------------------------------------------------------------------------
// launch
// ---------------------------------------------------------------------------
extern "C" void kernel_launch(void* const* d_in, const int* in_sizes, int n_in,
                              void* d_out, int out_size, void* d_ws, size_t ws_size,
                              hipStream_t stream) {
  (void)n_in; (void)out_size; (void)ws_size;

  const float* x    = (const float*)d_in[0];
  const int*   ei   = (const int*)d_in[1];
  const float* ew   = (const float*)d_in[2];
  const float* W1   = (const float*)d_in[3];
  const float* a1s  = (const float*)d_in[4];
  const float* a1d  = (const float*)d_in[5];
  const float* b1   = (const float*)d_in[6];
  const float* W2   = (const float*)d_in[7];
  const float* a2s  = (const float*)d_in[8];
  const float* a2d  = (const float*)d_in[9];
  const float* b2   = (const float*)d_in[10];

  const int Nn = in_sizes[0] / IN_F;
  const int E  = in_sizes[1] / 2;
  const int* src = ei;
  const int* dst = ei + E;
  const unsigned k = (unsigned)((double)E * 0.1);
  const int full_tiles = Nn / 16;
  const int tail_rows  = Nn - full_tiles * 16;

  // workspace carve-out
  char* ws = (char*)d_ws;
  size_t off = 0;
  auto alloc = [&](size_t bytes) -> void* {
    void* p = ws + off;
    off += (bytes + 255) & ~(size_t)255;
    return p;
  };
  float*    h1    = (float*)alloc((size_t)Nn * IN_F * 4);   // layer1 proj, reused as h2
  float*    agg1  = (float*)alloc((size_t)Nn * IN_F * 4);   // layer1 agg -> x2 in place
  float*    as    = (float*)alloc((size_t)Nn * HEADS * 4);
  float*    ad    = (float*)alloc((size_t)Nn * HEADS * 4);
  float*    mx    = (float*)alloc((size_t)Nn * HEADS * 4);
  float*    den   = (float*)alloc((size_t)Nn * HEADS * 4);
  __bf16*   Wp1   = (__bf16*)alloc((size_t)IN_F * IN_F * 2);   // packed B, layer 1
  __bf16*   Wp2   = (__bf16*)alloc((size_t)IN_F * OUT_F * 2);  // packed B, layer 2
  unsigned* hist  = (unsigned*)alloc(2048 * 4);
  unsigned* state = (unsigned*)alloc(4 * 4);
  float*    thr   = (float*)alloc(4);
  float*    out   = (float*)d_out;

  const int TB = 256;
  auto nb = [&](long long n) { return (int)((n + TB - 1) / TB); };

  // ---- exact threshold (k-th largest of edge weights) ----
  fill_u32<<<nb(2048), TB, 0, stream>>>(hist, 0u, 2048);
  thresh_hist<<<nb(E), TB, 0, stream>>>(ew, E, state, hist, 0);
  thresh_scan<<<1, 32, 0, stream>>>(hist, state, 2048, 0, k, thr);
  thresh_hist<<<nb(E), TB, 0, stream>>>(ew, E, state, hist, 1);
  thresh_scan<<<1, 32, 0, stream>>>(hist, state, 2048, 1, k, thr);
  thresh_hist<<<nb(E), TB, 0, stream>>>(ew, E, state, hist, 2);
  thresh_scan<<<1, 32, 0, stream>>>(hist, state, 1024, 2, k, thr);

  // ---- weight packing into WMMA B layout ----
  pack_B<IN_F><<<nb(IN_F * IN_F), TB, 0, stream>>>(W1, Wp1);
  pack_B<OUT_F><<<nb(IN_F * OUT_F), TB, 0, stream>>>(W2, Wp2);

  // ================= layer 1 (128 -> 4x32, concat) =================
  if (full_tiles > 0)
    gemm_wmma<IN_F><<<full_tiles, 2 * IN_F, 0, stream>>>(x, Wp1, h1);
  if (tail_rows > 0)
    gemm_tail<IN_F><<<nb(tail_rows * IN_F), TB, 0, stream>>>(x, W1, h1,
                                                             full_tiles * 16, tail_rows);
  alpha_kernel<HEADS, HIDC><<<nb((long long)Nn * HEADS), TB, 0, stream>>>(h1, a1s, a1d,
                                                                          as, ad, Nn);
  fill_f32<<<nb((long long)Nn * HEADS), TB, 0, stream>>>(mx, -3.0e38f, Nn * HEADS);
  fill_f32<<<nb((long long)Nn * HEADS), TB, 0, stream>>>(den, 0.f, Nn * HEADS);
  fill_f32<<<nb((long long)Nn * IN_F), TB, 0, stream>>>(agg1, 0.f, Nn * IN_F);

  long long t1 = (long long)(E + Nn) * HEADS;
  edge_score_kernel<HEADS><<<nb(t1), TB, 0, stream>>>(src, dst, ew, thr, as, ad, mx, den,
                                                      E, Nn, 0);
  edge_score_kernel<HEADS><<<nb(t1), TB, 0, stream>>>(src, dst, ew, thr, as, ad, mx, den,
                                                      E, Nn, 1);
  long long t1a = (long long)(E + Nn) * 32;
  edge_agg_kernel<HEADS, HIDC><<<nb(t1a), TB, 0, stream>>>(src, dst, ew, thr, as, ad,
                                                           mx, den, h1, agg1, E, Nn);
  relu_bias_kernel<<<nb((long long)Nn * IN_F), TB, 0, stream>>>(agg1, b1, Nn * IN_F, IN_F);

  // ================= layer 2 (128 -> 64, 1 head) =================
  if (full_tiles > 0)
    gemm_wmma<OUT_F><<<full_tiles, 2 * OUT_F, 0, stream>>>(agg1, Wp2, h1);
  if (tail_rows > 0)
    gemm_tail<OUT_F><<<nb(tail_rows * OUT_F), TB, 0, stream>>>(agg1, W2, h1,
                                                               full_tiles * 16, tail_rows);
  alpha_kernel<1, OUT_F><<<nb((long long)Nn), TB, 0, stream>>>(h1, a2s, a2d, as, ad, Nn);
  fill_f32<<<nb((long long)Nn), TB, 0, stream>>>(mx, -3.0e38f, Nn);
  fill_f32<<<nb((long long)Nn), TB, 0, stream>>>(den, 0.f, Nn);
  fill_f32<<<nb((long long)Nn * OUT_F), TB, 0, stream>>>(out, 0.f, Nn * OUT_F);

  long long t2 = (long long)(E + Nn);
  edge_score_kernel<1><<<nb(t2), TB, 0, stream>>>(src, dst, ew, thr, as, ad, mx, den,
                                                  E, Nn, 0);
  edge_score_kernel<1><<<nb(t2), TB, 0, stream>>>(src, dst, ew, thr, as, ad, mx, den,
                                                  E, Nn, 1);
  edge_agg_kernel<1, OUT_F><<<nb(t1a), TB, 0, stream>>>(src, dst, ew, thr, as, ad,
                                                        mx, den, h1, out, E, Nn);
  add_bias_kernel<<<nb((long long)Nn * OUT_F), TB, 0, stream>>>(out, b2, Nn * OUT_F, OUT_F);
}
